// DecoderLayer_11458972745759
// MI455X (gfx1250) — matrix operand order repaired and checked by reading, not compiled, not run on hardware
//
#include <hip/hip_runtime.h>
#include <stdint.h>

// ---------------------------------------------------------------------------
// MI455X (gfx1250) transformer decoder layer, wave32 + WMMA bf16.
// Compute-bound (~172 GFLOP over ~150MB): all GEMMs run on
// V_WMMA_F32_16X16X32_BF16. GEMM tiles are staged with
// GLOBAL_LOAD_ASYNC_TO_LDS_B128 (ASYNCcnt) so tile movement never touches
// VGPRs; flash-attention keeps the [B,H,T,S] score tensor out of HBM.
// ---------------------------------------------------------------------------

#define B_SZ 4
#define T_SZ 1024
#define S_SZ 1024
#define D_SZ 1024
#define H_SZ 16
#define DH_SZ 64
#define FF_SZ 4096

typedef __bf16 bf16_t;
typedef bf16_t v16bf __attribute__((ext_vector_type(16)));
typedef bf16_t v8bf  __attribute__((ext_vector_type(8)));
typedef float  v8f   __attribute__((ext_vector_type(8)));
typedef unsigned short u16;
typedef u16 v8u16 __attribute__((ext_vector_type(8)));

static __device__ __forceinline__ u16 f2bf_bits(float f) {
  unsigned int u = __builtin_bit_cast(unsigned int, f);
  unsigned int r = u + 0x7fffu + ((u >> 16) & 1u);
  return (u16)(r >> 16);
}

// A-operand fragment, 16x32 bf16 (ISA 7.12.2): lane L holds row m = L%16,
// K chunks [koff, koff+8) and [16+koff, 16+koff+8), koff = (L>=16)*8.
static __device__ __forceinline__ v16bf frag_a(const u16* p0, int ld) {
  const int lane = threadIdx.x & 31;
  const u16* p = p0 + (lane & 15) * ld + ((lane >> 4) << 3);
  v8bf lo = *(const v8bf*)(p);
  v8bf hi = *(const v8bf*)(p + 16);
  v16bf f;
#pragma unroll
  for (int i = 0; i < 8; ++i) { f[i] = lo[i]; f[i + 8] = hi[i]; }
  return f;
}

// B-operand fragment, 32x16 bf16: lane L holds column n = L%16, K chunks
// [koff, koff+8) and [16+koff, 16+koff+8), koff = (L>=16)*8. Fed with rows of W^T [N,K] / K[s,d] / V^T[d,s].
static __device__ __forceinline__ v16bf frag_b(const u16* p0, int ld) {
  const int lane = threadIdx.x & 31;
  const u16* p = p0 + (lane & 15) * ld + ((lane >> 4) << 3);
  v8bf lo = *(const v8bf*)(p);
  v8bf hi = *(const v8bf*)(p + 16);
  v16bf f;
#pragma unroll
  for (int i = 0; i < 8; ++i) { f[i] = lo[i]; f[i + 8] = hi[i]; }
  return f;
}

static __device__ __forceinline__ v8f wmma_bf16(v16bf a, v16bf b, v8f c) {
  return __builtin_amdgcn_wmma_f32_16x16x32_bf16(false, a, false, b,
                                                 (short)0, c, false, false);
}

// CDNA5 async global->LDS copy (GVS mode): per-lane 16B, ASYNCcnt-tracked.
// INST_OFFSET is added to BOTH the LDS and the global address (ISA 10.4.4).
template <int IOFF>
static __device__ __forceinline__ void async_ld_b128(unsigned lds_off,
                                                     unsigned gl_off,
                                                     const void* sbase) {
  asm volatile("global_load_async_to_lds_b128 %0, %1, %2 offset:%3"
               :
               : "v"(lds_off), "v"(gl_off), "s"(sbase), "n"(IOFF)
               : "memory");
}
static __device__ __forceinline__ void wait_async0() {
  asm volatile("s_wait_asynccnt 0x0" ::: "memory");
}

// -------------------------- elementwise helpers ----------------------------

__global__ __launch_bounds__(256) void k_cvt_bf16(const float* __restrict__ in,
                                                  u16* __restrict__ out, int n) {
  for (int i = blockIdx.x * 256 + threadIdx.x; i < n; i += gridDim.x * 256)
    out[i] = f2bf_bits(in[i]);
}

// in: [K][N] fp32 row-major  ->  out: [N][K] bf16 row-major (W^T)
__global__ __launch_bounds__(256) void k_cvt_transpose(const float* __restrict__ in,
                                                       u16* __restrict__ out,
                                                       int K, int N) {
  int total = K * N;
  for (int i = blockIdx.x * 256 + threadIdx.x; i < total; i += gridDim.x * 256) {
    int nn = i / K, kk = i - nn * K;
    out[i] = f2bf_bits(in[(size_t)kk * N + nn]);
  }
}

__global__ __launch_bounds__(256) void k_copy_f32(const float* __restrict__ in,
                                                  float* __restrict__ out, int n) {
  for (int i = blockIdx.x * 256 + threadIdx.x; i < n; i += gridDim.x * 256)
    out[i] = in[i];
}

// kv [B,S,2D] bf16 -> vt [B,H,DH,S] bf16 (V transposed per head)
__global__ __launch_bounds__(256) void k_repack_v(const u16* __restrict__ kv,
                                                  u16* __restrict__ vt) {
  int n = B_SZ * H_SZ * DH_SZ * S_SZ;
  for (int i = blockIdx.x * 256 + threadIdx.x; i < n; i += gridDim.x * 256) {
    int s = i & (S_SZ - 1);
    int d = (i >> 10) & (DH_SZ - 1);
    int h = (i >> 16) & (H_SZ - 1);
    int b = i >> 20;
    vt[i] = kv[((size_t)(b * S_SZ + s)) * (2 * D_SZ) + D_SZ + h * DH_SZ + d];
  }
}

// ------------------------------ tiled GEMM ---------------------------------
// C[M,N] = act(scale * (A_bf16[M,K] @ Wt_bf16[N,K]^T + bias)).
// 128x128 block, BK=32, 8 waves in 4(M)x2(N); each wave 32x64 (8 WMMA per
// k-step from 6 fragment loads). Tiles staged by async global->LDS copies,
// double buffered; K-loop tail peeled so the steady state is branch-free.

#define LDP 40  // 32 + 8 halves pad: keeps 16B alignment, breaks bank cycle
#define BMB 128
#define BNB 128

__global__ __launch_bounds__(256)
void k_gemm_bf16(const u16* __restrict__ A, int lda,
                 const u16* __restrict__ Wt,
                 const float* __restrict__ bias,
                 float* __restrict__ Cf, u16* __restrict__ Cb,
                 int M, int N, int K, float scale, int relu) {
  __shared__ u16 As[2][BMB * LDP];
  __shared__ u16 Bs[2][BNB * LDP];

  const int tid = threadIdx.x;
  const int lane = tid & 31;
  const int wid = tid >> 5;
  const int wm = wid >> 1;   // 0..3
  const int wn = wid & 1;    // 0..1
  const int m0 = blockIdx.y * BMB;
  const int n0 = blockIdx.x * BNB;

  // loaders: each thread moves 16 el (2 x b128) of A and of B per k-step
  const int lr = tid >> 1, lc = (tid & 1) * 16;
  const unsigned goffA = (unsigned)(((size_t)(m0 + lr) * lda + lc) * 2);
  const unsigned goffB = (unsigned)(((size_t)(n0 + lr) * K + lc) * 2);
  unsigned ldsA[2], ldsB[2];
  ldsA[0] = (unsigned)(size_t)&As[0][lr * LDP + lc];
  ldsA[1] = (unsigned)(size_t)&As[1][lr * LDP + lc];
  ldsB[0] = (unsigned)(size_t)&Bs[0][lr * LDP + lc];
  ldsB[1] = (unsigned)(size_t)&Bs[1][lr * LDP + lc];

  v8f acc[2][4] = {};

#define ISSUE_TILE(buf, k0)                                                   \
  do {                                                                        \
    const char* sa = (const char*)A + (size_t)(k0) * 2;                       \
    const char* sb = (const char*)Wt + (size_t)(k0) * 2;                      \
    async_ld_b128<0>(ldsA[buf], goffA, sa);                                   \
    async_ld_b128<16>(ldsA[buf], goffA, sa);                                  \
    async_ld_b128<0>(ldsB[buf], goffB, sb);                                   \
    async_ld_b128<16>(ldsB[buf], goffB, sb);                                  \
  } while (0)

#define COMPUTE_TILE(buf)                                                     \
  do {                                                                        \
    const u16* ab = &As[buf][(wm * 32) * LDP];                                \
    const u16* bb = &Bs[buf][(wn * 64) * LDP];                                \
    v16bf a0 = frag_a(ab, LDP);                                               \
    v16bf a1 = frag_a(ab + 16 * LDP, LDP);                                    \
    v16bf b0 = frag_b(bb, LDP);                                               \
    v16bf b1 = frag_b(bb + 16 * LDP, LDP);                                    \
    v16bf b2 = frag_b(bb + 32 * LDP, LDP);                                    \
    v16bf b3 = frag_b(bb + 48 * LDP, LDP);                                    \
    acc[0][0] = wmma_bf16(a0, b0, acc[0][0]);                                 \
    acc[0][1] = wmma_bf16(a0, b1, acc[0][1]);                                 \
    acc[0][2] = wmma_bf16(a0, b2, acc[0][2]);                                 \
    acc[0][3] = wmma_bf16(a0, b3, acc[0][3]);                                 \
    acc[1][0] = wmma_bf16(a1, b0, acc[1][0]);                                 \
    acc[1][1] = wmma_bf16(a1, b1, acc[1][1]);                                 \
    acc[1][2] = wmma_bf16(a1, b2, acc[1][2]);                                 \
    acc[1][3] = wmma_bf16(a1, b3, acc[1][3]);                                 \
  } while (0)

  ISSUE_TILE(0, 0);
  int cur = 0;
  for (int k0 = 0; k0 < K - 32; k0 += 32) {
    wait_async0();
    __syncthreads();
    ISSUE_TILE(cur ^ 1, k0 + 32);
    COMPUTE_TILE(cur);
    cur ^= 1;
  }
  wait_async0();
  __syncthreads();
  COMPUTE_TILE(cur);
#undef ISSUE_TILE
#undef COMPUTE_TILE

  const int ncol = lane & 15;
  const int rbase = (lane >> 4) << 3;  // C layout: rows rbase..rbase+7
#pragma unroll
  for (int i = 0; i < 2; ++i) {
#pragma unroll
    for (int j = 0; j < 4; ++j) {
      const int mm = m0 + wm * 32 + i * 16 + rbase;
      const int nn = n0 + wn * 64 + j * 16 + ncol;
      const float bv = bias ? bias[nn] : 0.0f;
#pragma unroll
      for (int r = 0; r < 8; ++r) {
        float v = (acc[i][j][r] + bv) * scale;
        if (relu) v = fmaxf(v, 0.0f);
        if (Cf) Cf[(size_t)(mm + r) * N + nn] = v;
        if (Cb) Cb[(size_t)(mm + r) * N + nn] = f2bf_bits(v);
      }
    }
  }
}

// --------------------------- flash attention -------------------------------
// One wave per (b, h, 16 Q rows). Q pre-scaled by dh^-0.5 in its projection.
// kv: [B,S,2D] bf16 (K = first D cols). vt: [B,H,DH,S] bf16. mask: bool bytes.

#define NEG_BIG (-1e30f)

__global__ __launch_bounds__(32)
void k_flash_attn(const u16* __restrict__ Q, const u16* __restrict__ KV,
                  const u16* __restrict__ VT, const unsigned char* __restrict__ mask,
                  u16* __restrict__ CTX) {
  const int lane = threadIdx.x;
  const int mt = blockIdx.x, h = blockIdx.y, b = blockIdx.z;
  __shared__ u16 P[16 * LDP];

  const u16* qb = Q + ((size_t)(b * T_SZ) + mt * 16) * D_SZ + h * DH_SZ;
  const v16bf aq0 = frag_a(qb, D_SZ);
  const v16bf aq1 = frag_a(qb + 32, D_SZ);

  v8f acc[4] = {};
  float mrow[8], lrow[8];
#pragma unroll
  for (int r = 0; r < 8; ++r) { mrow[r] = NEG_BIG; lrow[r] = 0.0f; }

  const int ncol = lane & 15;
  const int rbase = (lane >> 4) << 3;
  const unsigned char* mbase =
      mask + ((size_t)b * T_SZ + mt * 16 + rbase) * S_SZ;
  const u16* kb = KV + (size_t)b * S_SZ * (2 * D_SZ) + h * DH_SZ;
  const u16* vb = VT + ((size_t)(b * H_SZ) + h) * DH_SZ * S_SZ;

  for (int sc = 0; sc < S_SZ; sc += 32) {
    // ---- scores: [16 x 32] = Q[16x64] @ K[sc..sc+32, 64]^T
    v8f s0 = {}, s1 = {};
    s0 = wmma_bf16(aq0, frag_b(kb + (size_t)sc * (2 * D_SZ), 2 * D_SZ), s0);
    s0 = wmma_bf16(aq1, frag_b(kb + (size_t)sc * (2 * D_SZ) + 32, 2 * D_SZ), s0);
    s1 = wmma_bf16(aq0, frag_b(kb + (size_t)(sc + 16) * (2 * D_SZ), 2 * D_SZ), s1);
    s1 = wmma_bf16(aq1, frag_b(kb + (size_t)(sc + 16) * (2 * D_SZ) + 32, 2 * D_SZ), s1);

    // ---- mask + online softmax (rows rbase..rbase+7, col = sc + ncol [+16])
#pragma unroll
    for (int r = 0; r < 8; ++r) {
      const unsigned char* mr = mbase + (size_t)r * S_SZ + sc + ncol;
      if (mr[0])  s0[r] = NEG_BIG;
      if (mr[16]) s1[r] = NEG_BIG;
    }
#pragma unroll
    for (int r = 0; r < 8; ++r) {
      float cm = fmaxf(s0[r], s1[r]);
      cm = fmaxf(cm, __shfl_xor(cm, 1, 32));
      cm = fmaxf(cm, __shfl_xor(cm, 2, 32));
      cm = fmaxf(cm, __shfl_xor(cm, 4, 32));
      cm = fmaxf(cm, __shfl_xor(cm, 8, 32));
      const float nm = fmaxf(mrow[r], cm);
      const float alpha = __expf(mrow[r] - nm);
      mrow[r] = nm;
      const float p0 = __expf(s0[r] - nm);
      const float p1 = __expf(s1[r] - nm);
      s0[r] = p0; s1[r] = p1;
      float ps = p0 + p1;
      ps += __shfl_xor(ps, 1, 32);
      ps += __shfl_xor(ps, 2, 32);
      ps += __shfl_xor(ps, 4, 32);
      ps += __shfl_xor(ps, 8, 32);
      lrow[r] = lrow[r] * alpha + ps;
      acc[0][r] *= alpha; acc[1][r] *= alpha;
      acc[2][r] *= alpha; acc[3][r] *= alpha;
    }

    // ---- relayout P (C-layout) -> A-layout via LDS
    __syncthreads();
#pragma unroll
    for (int r = 0; r < 8; ++r) {
      P[(rbase + r) * LDP + ncol]      = f2bf_bits(s0[r]);
      P[(rbase + r) * LDP + 16 + ncol] = f2bf_bits(s1[r]);
    }
    __syncthreads();
    const v16bf ap = frag_a(P, LDP);

    // ---- ctx += P[16x32] @ V[sc..sc+32, 64]  (B-operand rows from V^T)
#pragma unroll
    for (int j = 0; j < 4; ++j)
      acc[j] = wmma_bf16(ap, frag_b(vb + (size_t)(j * 16) * S_SZ + sc, S_SZ), acc[j]);
  }

  // ---- normalize and write ctx (bf16, [B,T,D] with head offset)
  u16* cb = CTX + ((size_t)(b * T_SZ) + mt * 16 + rbase) * D_SZ + h * DH_SZ + ncol;
#pragma unroll
  for (int r = 0; r < 8; ++r) {
    const float inv = 1.0f / lrow[r];
#pragma unroll
    for (int j = 0; j < 4; ++j)
      cb[(size_t)r * D_SZ + j * 16] = f2bf_bits(acc[j][r] * inv);
  }
}

// ----------------------- residual + layernorm ------------------------------

__global__ __launch_bounds__(256)
void k_ln_res(const float* __restrict__ A, const float* __restrict__ Bv,
              const float* __restrict__ g, const float* __restrict__ be,
              float* __restrict__ outF, u16* __restrict__ outB) {
  const int row = blockIdx.x, tid = threadIdx.x;
  const float* a = A + (size_t)row * D_SZ;
  const float* b = Bv + (size_t)row * D_SZ;
  float v[4], s1 = 0.0f, s2 = 0.0f;
#pragma unroll
  for (int i = 0; i < 4; ++i) {
    float t = a[tid + i * 256] + b[tid + i * 256];
    v[i] = t; s1 += t; s2 += t * t;
  }
  __shared__ float r1[256], r2[256];
  r1[tid] = s1; r2[tid] = s2;
  __syncthreads();
  for (int off = 128; off > 0; off >>= 1) {
    if (tid < off) { r1[tid] += r1[tid + off]; r2[tid] += r2[tid + off]; }
    __syncthreads();
  }
  const float mean = r1[0] * (1.0f / D_SZ);
  const float var = r2[0] * (1.0f / D_SZ) - mean * mean;
  const float rstd = rsqrtf(var + 1e-5f);
#pragma unroll
  for (int i = 0; i < 4; ++i) {
    const int c = tid + i * 256;
    const float y = (v[i] - mean) * rstd * g[c] + be[c];
    outF[(size_t)row * D_SZ + c] = y;
    if (outB) outB[(size_t)row * D_SZ + c] = f2bf_bits(y);
  }
}

// ------------------------------- driver ------------------------------------

extern "C" void kernel_launch(void* const* d_in, const int* in_sizes, int n_in,
                              void* d_out, int out_size, void* d_ws, size_t ws_size,
                              hipStream_t stream) {
  (void)in_sizes; (void)n_in; (void)out_size; (void)ws_size;
  const float* x    = (const float*)d_in[0];
  const float* enc  = (const float*)d_in[1];
  const unsigned char* src_mask = (const unsigned char*)d_in[2];
  const unsigned char* tgt_mask = (const unsigned char*)d_in[3];
  const float* wq1 = (const float*)d_in[4];  const float* bq1 = (const float*)d_in[5];
  const float* wkv1= (const float*)d_in[6];  const float* bkv1= (const float*)d_in[7];
  const float* wo1 = (const float*)d_in[8];  const float* bo1 = (const float*)d_in[9];
  const float* wq2 = (const float*)d_in[10]; const float* bq2 = (const float*)d_in[11];
  const float* wkv2= (const float*)d_in[12]; const float* bkv2= (const float*)d_in[13];
  const float* wo2 = (const float*)d_in[14]; const float* bo2 = (const float*)d_in[15];
  const float* w_in = (const float*)d_in[16]; const float* b_in = (const float*)d_in[17];
  const float* w_out= (const float*)d_in[18]; const float* b_out= (const float*)d_in[19];
  const float* g0 = (const float*)d_in[20]; const float* be0 = (const float*)d_in[21];
  const float* g1 = (const float*)d_in[22]; const float* be1 = (const float*)d_in[23];
  const float* g2 = (const float*)d_in[24]; const float* be2 = (const float*)d_in[25];

  char* wp = (char*)d_ws;
  auto alloc = [&](size_t bytes) -> void* {
    void* r = (void*)wp;
    wp += (bytes + 255) & ~(size_t)255;
    return r;
  };
  const size_t BT = (size_t)B_SZ * T_SZ;          // 4096 rows
  u16* wq1t  = (u16*)alloc((size_t)D_SZ * D_SZ * 2);
  u16* wkv1t = (u16*)alloc((size_t)2 * D_SZ * D_SZ * 2);
  u16* wo1t  = (u16*)alloc((size_t)D_SZ * D_SZ * 2);
  u16* wq2t  = (u16*)alloc((size_t)D_SZ * D_SZ * 2);
  u16* wkv2t = (u16*)alloc((size_t)2 * D_SZ * D_SZ * 2);
  u16* wo2t  = (u16*)alloc((size_t)D_SZ * D_SZ * 2);
  u16* w_int = (u16*)alloc((size_t)FF_SZ * D_SZ * 2);
  u16* w_outt= (u16*)alloc((size_t)D_SZ * FF_SZ * 2);
  u16* x_bf  = (u16*)alloc(BT * D_SZ * 2);
  u16* enc_bf= (u16*)alloc(BT * D_SZ * 2);
  u16* q_bf  = (u16*)alloc(BT * D_SZ * 2);
  u16* kv_bf = (u16*)alloc(BT * 2 * D_SZ * 2);
  u16* vt    = (u16*)alloc(BT * D_SZ * 2);
  u16* ctx_bf= (u16*)alloc(BT * D_SZ * 2);
  float* ybuf= (float*)alloc(BT * D_SZ * 4);
  float* x1  = (float*)alloc(BT * D_SZ * 4);
  u16* x1_bf = (u16*)alloc(BT * D_SZ * 2);
  float* x2  = (float*)alloc(BT * D_SZ * 4);
  u16* x2_bf = (u16*)alloc(BT * D_SZ * 2);
  u16* h_bf  = (u16*)alloc(BT * FF_SZ * 2);

  const int EW = 2048;  // elementwise grid blocks
  // weight transposes (fp32 [K][N] -> bf16 [N][K])
  k_cvt_transpose<<<EW, 256, 0, stream>>>(wq1,  wq1t,  D_SZ, D_SZ);
  k_cvt_transpose<<<EW, 256, 0, stream>>>(wkv1, wkv1t, D_SZ, 2 * D_SZ);
  k_cvt_transpose<<<EW, 256, 0, stream>>>(wo1,  wo1t,  D_SZ, D_SZ);
  k_cvt_transpose<<<EW, 256, 0, stream>>>(wq2,  wq2t,  D_SZ, D_SZ);
  k_cvt_transpose<<<EW, 256, 0, stream>>>(wkv2, wkv2t, D_SZ, 2 * D_SZ);
  k_cvt_transpose<<<EW, 256, 0, stream>>>(wo2,  wo2t,  D_SZ, D_SZ);
  k_cvt_transpose<<<EW, 256, 0, stream>>>(w_in, w_int, D_SZ, FF_SZ);
  k_cvt_transpose<<<EW, 256, 0, stream>>>(w_out, w_outt, FF_SZ, D_SZ);
  k_cvt_bf16<<<EW, 256, 0, stream>>>(x,   x_bf,   (int)(BT * D_SZ));
  k_cvt_bf16<<<EW, 256, 0, stream>>>(enc, enc_bf, (int)(BT * D_SZ));

  const float qscale = 0.125f;  // dh^-0.5, dh=64
  dim3 gD(D_SZ / BNB, BT / BMB), g2D(2 * D_SZ / BNB, BT / BMB), gFF(FF_SZ / BNB, BT / BMB);
  dim3 gAtt(T_SZ / 16, H_SZ, B_SZ);

  // ---- self attention ----
  k_gemm_bf16<<<gD, 256, 0, stream>>>(x_bf, D_SZ, wq1t, bq1, nullptr, q_bf,
                                      (int)BT, D_SZ, D_SZ, qscale, 0);
  k_gemm_bf16<<<g2D, 256, 0, stream>>>(x_bf, D_SZ, wkv1t, bkv1, nullptr, kv_bf,
                                       (int)BT, 2 * D_SZ, D_SZ, 1.0f, 0);
  k_repack_v<<<EW, 256, 0, stream>>>(kv_bf, vt);
  k_flash_attn<<<gAtt, 32, 0, stream>>>(q_bf, kv_bf, vt, tgt_mask, ctx_bf);
  k_gemm_bf16<<<gD, 256, 0, stream>>>(ctx_bf, D_SZ, wo1t, bo1, ybuf, nullptr,
                                      (int)BT, D_SZ, D_SZ, 1.0f, 0);
  k_ln_res<<<(int)BT, 256, 0, stream>>>(x, ybuf, g0, be0, x1, x1_bf);

  // ---- cross attention ----
  k_gemm_bf16<<<gD, 256, 0, stream>>>(x1_bf, D_SZ, wq2t, bq2, nullptr, q_bf,
                                      (int)BT, D_SZ, D_SZ, qscale, 0);
  k_gemm_bf16<<<g2D, 256, 0, stream>>>(enc_bf, D_SZ, wkv2t, bkv2, nullptr, kv_bf,
                                       (int)BT, 2 * D_SZ, D_SZ, 1.0f, 0);
  k_repack_v<<<EW, 256, 0, stream>>>(kv_bf, vt);
  k_flash_attn<<<gAtt, 32, 0, stream>>>(q_bf, kv_bf, vt, src_mask, ctx_bf);
  k_gemm_bf16<<<gD, 256, 0, stream>>>(ctx_bf, D_SZ, wo2t, bo2, ybuf, nullptr,
                                      (int)BT, D_SZ, D_SZ, 1.0f, 0);
  k_ln_res<<<(int)BT, 256, 0, stream>>>(x1, ybuf, g1, be1, x2, x2_bf);

  // ---- feed forward ----
  k_gemm_bf16<<<gFF, 256, 0, stream>>>(x2_bf, D_SZ, w_int, b_in, nullptr, h_bf,
                                       (int)BT, FF_SZ, D_SZ, 1.0f, 1 /*relu*/);
  k_gemm_bf16<<<gD, 256, 0, stream>>>(h_bf, FF_SZ, w_outt, b_out, ybuf, nullptr,
                                      (int)BT, D_SZ, FF_SZ, 1.0f, 0);

  // ---- final LN -> d_out[0 : B*T*D], then all_input -> d_out[B*T*D : ] ----
  float* out0 = (float*)d_out;
  k_ln_res<<<(int)BT, 256, 0, stream>>>(x2, ybuf, g2, be2, out0, nullptr);
  k_copy_f32<<<EW, 256, 0, stream>>>(x, out0 + BT * D_SZ, (int)(BT * D_SZ));
}
